// MultiHeadAttention_21930103013951
// MI455X (gfx1250) — matrix operand order repaired and checked by reading, not compiled, not run on hardware
//
#include <hip/hip_runtime.h>

// ---------------------------------------------------------------------------
// MHA forward for MI455X (gfx1250), bf16 WMMA with f32 accumulation.
//   B=2, T=2048, C=1024, H=16, Dk=64
// Pipeline:
//   (1) f32->bf16 convert of activations, transpose+convert of weights
//   (2) QKV projections: bf16 WMMA GEMM (32x64 tile/wave), bf16 output
//   (3) flash-style attention: WMMA QK^T + online softmax + WMMA PV, bf16 out
//   (4) output projection: bf16 WMMA GEMM, f32 output
// ---------------------------------------------------------------------------

typedef __attribute__((ext_vector_type(16))) __bf16 v16bf;
typedef __attribute__((ext_vector_type(8)))  __bf16 v8bf;
typedef __attribute__((ext_vector_type(8)))  float  v8f;

#define WMMA_BF16(a, b, c) \
  __builtin_amdgcn_wmma_f32_16x16x32_bf16(false, (a), false, (b), (short)0, (c), false, false)

static constexpr int kB  = 2;
static constexpr int kT  = 2048;
static constexpr int kC  = 1024;
static constexpr int kH  = 16;
static constexpr int kDK = 64;
static constexpr int kM  = kB * kT;   // 4096 rows in all GEMMs

// f32 -> bf16 round-to-nearest-even via bit ops (no reliance on __bf16 casts)
static __device__ __forceinline__ __bf16 f2bf(float x) {
  unsigned u = __builtin_bit_cast(unsigned, x);
  unsigned r = u + 0x7FFFu + ((u >> 16) & 1u);
  unsigned short h = (unsigned short)(r >> 16);
  return __builtin_bit_cast(__bf16, h);
}

static __device__ __forceinline__ v16bf cat8(v8bf lo, v8bf hi) {
  return __builtin_shufflevector(lo, hi, 0,1,2,3,4,5,6,7,8,9,10,11,12,13,14,15);
}

// ---------------------------------------------------------------------------
// Elementwise f32 -> bf16, 8 elements per thread (32B load / 16B store).
// ---------------------------------------------------------------------------
__global__ __launch_bounds__(256) void convert_bf16(
    const float* __restrict__ in, __bf16* __restrict__ out) {
  const int idx = (blockIdx.x * 256 + threadIdx.x) * 8;
  v8f f = *(const v8f*)(in + idx);
  v8bf o;
#pragma unroll
  for (int i = 0; i < 8; ++i) o[i] = f2bf(f[i]);
  *(v8bf*)(out + idx) = o;
}

// ---------------------------------------------------------------------------
// Weight prep: W [K=1024, N=1024] f32 (row-major [in,out]) -> WT [N,K] bf16
// so GEMM B-fragments are K-contiguous.
// ---------------------------------------------------------------------------
__global__ __launch_bounds__(256) void transpose_convert_1024(
    const float* __restrict__ in, __bf16* __restrict__ out) {
  int idx = blockIdx.x * 256 + threadIdx.x;   // 0 .. 1M-1
  int n = idx >> 10;
  int k = idx & 1023;
  out[idx] = f2bf(in[k * 1024 + n]);          // writes coalesced over k
}

// ---------------------------------------------------------------------------
// GEMM: C[4096,1024] = A[4096,1024](bf16) * BT[1024,1024](bf16)^T + bias
// One wave computes a 32x64 tile: 2 A-frags x 4 B-frags -> 8 WMMA per k-step.
// Dims hardcoded -> pure shift/mask index math.
// ---------------------------------------------------------------------------
template <bool OUT_BF16>
__global__ __launch_bounds__(256) void gemm_bf16_wmma(
    const __bf16* __restrict__ A, const __bf16* __restrict__ BT,
    const float* __restrict__ bias,
    __bf16* __restrict__ outb, float* __restrict__ outf) {
  constexpr int N = kC, K = kC;
  const int lane = threadIdx.x & 31;
  const int wave = threadIdx.x >> 5;
  const int wid  = blockIdx.x * 8 + wave;    // 0..2047
  const int m0 = (wid & 127) << 5;           // 128 M-tiles of 32
  const int n0 = (wid >> 7) << 6;            // 16 N-groups of 64

  const int hi = lane >> 4;                  // half-wave select
  const int ln = lane & 15;
  const int akb = hi << 4;                   // A k-base: 0 or 16

  const __bf16* aptr0 = A + (size_t)(m0 + ln) * K + akb;
  const __bf16* aptr1 = aptr0 + (size_t)16 * K;

  v8f acc[2][4] = {};

#pragma unroll 2
  for (int k0 = 0; k0 < K; k0 += 32) {
    // A fragments: row = m0 + 16*mm + lane%16, K = akb+{0..7} and akb+8+{0..7}
    v16bf a0 = cat8(*(const v8bf*)(aptr0 + k0), *(const v8bf*)(aptr0 + k0 + 8));
    v16bf a1 = cat8(*(const v8bf*)(aptr1 + k0), *(const v8bf*)(aptr1 + k0 + 8));

    const int kb = k0 + (hi << 4);           // B k-base: 16*(lane/16)
#pragma unroll
    for (int t = 0; t < 4; ++t) {
      const __bf16* bp = BT + (size_t)(n0 + 16 * t + ln) * K + kb;
      v16bf b = *(const v16bf*)bp;           // 16 contiguous bf16 = 32B
      acc[0][t] = WMMA_BF16(a0, b, acc[0][t]);
      acc[1][t] = WMMA_BF16(a1, b, acc[1][t]);
    }
  }

  // Epilogue: lane holds rows m0 + 16*mm + r + 8*(lane/16), col n0 + 16t + ln
#pragma unroll
  for (int mm = 0; mm < 2; ++mm) {
#pragma unroll
    for (int t = 0; t < 4; ++t) {
      const int col = n0 + 16 * t + ln;
      const float bv = bias[col];
      union { v8f v; float f[8]; } u; u.v = acc[mm][t];
#pragma unroll
      for (int r = 0; r < 8; ++r) {
        const size_t row = (size_t)(m0 + 16 * mm + r + (hi << 3));
        const float val = u.f[r] + bv;
        if constexpr (OUT_BF16) outb[row * N + col] = f2bf(val);
        else                    outf[row * N + col] = val;
      }
    }
  }
}

// ---------------------------------------------------------------------------
// Flash-style attention. Block = (b, h, 128 query rows); 8 waves x 16 rows.
// Per 32-key chunk: stage K and V^T in LDS, S = Q K^T (4 wmma), online
// softmax (raw-score running max, exp2 with folded scale*log2e), P staged via
// per-wave LDS tile, O += P V (4 wmma). Output bf16.
// ---------------------------------------------------------------------------
__global__ __launch_bounds__(256) void attn_wmma(
    const __bf16* __restrict__ Qb, const __bf16* __restrict__ Kb,
    const __bf16* __restrict__ Vb, __bf16* __restrict__ Ob) {
  __shared__ __align__(32) __bf16 Kt[32 * 64];        // [key][d]
  __shared__ __align__(32) __bf16 VT[64 * 32];        // [d][key]
  __shared__ __align__(32) __bf16 Pt[8 * 16 * 32];    // per-wave [q][key]

  const int tid  = threadIdx.x;
  const int lane = tid & 31;
  const int wave = tid >> 5;
  const int hi = lane >> 4;
  const int ln = lane & 15;

  const int nQB = kT / 128;                 // 16
  const int qb  = blockIdx.x & (nQB - 1);
  const int bh  = blockIdx.x / nQB;
  const int b   = bh >> 4;                  // /kH
  const int h   = bh & 15;
  const int q0  = qb * 128 + wave * 16;

  const size_t headoff = (size_t)h * kDK;
  // softmax is shift-invariant: track max in raw-score units and fold
  // (1/sqrt(Dk)) * log2(e) into a single multiply inside exp2.
  const float c = 0.125f * 1.44269504088896340736f;

  // Q fragments, held in registers for the whole key loop.
  const __bf16* qrow = Qb + (size_t)(b * kT + q0 + ln) * kC + headoff;
  const int akb = hi << 4;
  v16bf aQ0 = cat8(*(const v8bf*)(qrow + akb),      *(const v8bf*)(qrow + akb + 8));
  v16bf aQ1 = cat8(*(const v8bf*)(qrow + 32 + akb), *(const v8bf*)(qrow + 32 + akb + 8));

  float mrow[8], lrow[8];
#pragma unroll
  for (int r = 0; r < 8; ++r) { mrow[r] = -__builtin_inff(); lrow[r] = 0.0f; }
  v8f o[4] = {};

  // Cooperative staging indices: 256 threads x 8 bf16 = 32x64 tile.
  const int crow = tid >> 3;                // 0..31 (key)
  const int ccol = (tid & 7) << 3;          // 0,8,..,56 (d)
  const size_t gbase = (size_t)(b * kT) * kC + headoff + ccol;

  __bf16* P = &Pt[wave * (16 * 32)];

  for (int j = 0; j < kT; j += 32) {
    __syncthreads();  // previous iteration's LDS readers done

    // Stage K row-major, V transposed.
    const size_t roff = gbase + (size_t)(j + crow) * kC;
    v8bf kv = *(const v8bf*)(Kb + roff);
    v8bf vv = *(const v8bf*)(Vb + roff);
    *(v8bf*)&Kt[crow * 64 + ccol] = kv;
#pragma unroll
    for (int i = 0; i < 8; ++i) VT[(ccol + i) * 32 + crow] = vv[i];

    if (j + 32 < kT) {  // hint the next chunk toward the caches
      __builtin_prefetch(Kb + roff + 32 * kC, 0, 1);
      __builtin_prefetch(Vb + roff + 32 * kC, 0, 1);
    }
    __syncthreads();

    // S = Q K^T : two 16x16 key groups, K-dim 64 = 2 wmma each.
    v8f s[2] = {};
#pragma unroll
    for (int g = 0; g < 2; ++g) {
      const int kr = g * 16 + ln;   // key row in LDS
      v16bf bK0 = *(const v16bf*)&Kt[kr * 64 + (hi << 4)];
      v16bf bK1 = *(const v16bf*)&Kt[kr * 64 + 32 + (hi << 4)];
      s[g] = WMMA_BF16(aQ0, bK0, s[g]);
      s[g] = WMMA_BF16(aQ1, bK1, s[g]);
    }

    // Online softmax over this 32-key chunk (raw-score units).
    union { v8f v; float f[8]; } s0, s1; s0.v = s[0]; s1.v = s[1];
#pragma unroll
    for (int r = 0; r < 8; ++r) {
      float cm = fmaxf(s0.f[r], s1.f[r]);
#pragma unroll
      for (int msk = 1; msk < 16; msk <<= 1) cm = fmaxf(cm, __shfl_xor(cm, msk));
      const float mnew = fmaxf(mrow[r], cm);
      const float alpha = exp2f((mrow[r] - mnew) * c);
      mrow[r] = mnew;
      const float p0 = exp2f((s0.f[r] - mnew) * c);
      const float p1 = exp2f((s1.f[r] - mnew) * c);
      float rs = p0 + p1;
#pragma unroll
      for (int msk = 1; msk < 16; msk <<= 1) rs += __shfl_xor(rs, msk);
      lrow[r] = lrow[r] * alpha + rs;
#pragma unroll
      for (int t = 0; t < 4; ++t) o[t][r] *= alpha;
      // P tile (row q, col key), C-layout scatter
      P[(r + hi * 8) * 32 + ln]      = f2bf(p0);
      P[(r + hi * 8) * 32 + 16 + ln] = f2bf(p1);
    }
    __syncthreads();  // order P stores / keep block uniform

    // O += P * V : P as 16x32 A fragment, V^T rows give K-contiguous B frags.
    v16bf aP = cat8(*(const v8bf*)&P[ln * 32 + akb],
                    *(const v8bf*)&P[ln * 32 + akb + 8]);
#pragma unroll
    for (int t = 0; t < 4; ++t) {
      v16bf bV = *(const v16bf*)&VT[(16 * t + ln) * 32 + (hi << 4)];
      o[t] = WMMA_BF16(aP, bV, o[t]);
    }
  }

  // Normalize and write O (bf16) back in [B*T, C] layout.
#pragma unroll
  for (int t = 0; t < 4; ++t) {
    union { v8f v; float f[8]; } u; u.v = o[t];
#pragma unroll
    for (int r = 0; r < 8; ++r) {
      const size_t row = (size_t)(b * kT + q0 + r + (hi << 3));
      const int col = (int)headoff + 16 * t + ln;
      Ob[row * kC + col] = f2bf(u.f[r] / lrow[r]);
    }
  }
}

// ---------------------------------------------------------------------------
// Host-side orchestration (graph-capture safe: only kernel launches).
// ---------------------------------------------------------------------------
extern "C" void kernel_launch(void* const* d_in, const int* in_sizes, int n_in,
                              void* d_out, int out_size, void* d_ws, size_t ws_size,
                              hipStream_t stream) {
  const float* q_in = (const float*)d_in[0];
  const float* k_in = (const float*)d_in[1];
  const float* v_in = (const float*)d_in[2];
  const float* Wq   = (const float*)d_in[3];
  const float* bq   = (const float*)d_in[4];
  const float* Wk   = (const float*)d_in[5];
  const float* bk   = (const float*)d_in[6];
  const float* Wv   = (const float*)d_in[7];
  const float* bv   = (const float*)d_in[8];
  const float* Wo   = (const float*)d_in[9];
  const float* bo   = (const float*)d_in[10];
  float* out = (float*)d_out;

  char* ws = (char*)d_ws;
  const size_t MB = 1ull << 20;
  __bf16* WqT = (__bf16*)(ws + 0 * MB);    // 4 x 2MB
  __bf16* WkT = (__bf16*)(ws + 2 * MB);
  __bf16* WvT = (__bf16*)(ws + 4 * MB);
  __bf16* WoT = (__bf16*)(ws + 6 * MB);
  __bf16* Xq  = (__bf16*)(ws + 8 * MB);    // bf16 activations, 8MB each
  __bf16* Xk  = (__bf16*)(ws + 16 * MB);
  __bf16* Xv  = (__bf16*)(ws + 24 * MB);
  __bf16* Qb  = (__bf16*)(ws + 32 * MB);
  __bf16* Kb  = (__bf16*)(ws + 40 * MB);
  __bf16* Vb  = (__bf16*)(ws + 48 * MB);
  __bf16* Ob  = (__bf16*)(ws + 56 * MB);

  // 1) convert activations, transpose+convert weights to bf16
  const int conv_blocks = (kM * kC) / (256 * 8);    // 2048
  convert_bf16<<<conv_blocks, 256, 0, stream>>>(q_in, Xq);
  convert_bf16<<<conv_blocks, 256, 0, stream>>>(k_in, Xk);
  convert_bf16<<<conv_blocks, 256, 0, stream>>>(v_in, Xv);
  transpose_convert_1024<<<4096, 256, 0, stream>>>(Wq, WqT);
  transpose_convert_1024<<<4096, 256, 0, stream>>>(Wk, WkT);
  transpose_convert_1024<<<4096, 256, 0, stream>>>(Wv, WvT);
  transpose_convert_1024<<<4096, 256, 0, stream>>>(Wo, WoT);

  // 2) QKV projections (bf16 outputs); 32x64 tile/wave -> 2048 waves
  const int gemm_blocks = (kM / 32) * (kC / 64) / 8;  // 256
  gemm_bf16_wmma<true><<<gemm_blocks, 256, 0, stream>>>(Xq, WqT, bq, Qb, nullptr);
  gemm_bf16_wmma<true><<<gemm_blocks, 256, 0, stream>>>(Xk, WkT, bk, Kb, nullptr);
  gemm_bf16_wmma<true><<<gemm_blocks, 256, 0, stream>>>(Xv, WvT, bv, Vb, nullptr);

  // 3) attention (bf16 O)
  const int attn_blocks = kB * kH * (kT / 128);       // 512
  attn_wmma<<<attn_blocks, 256, 0, stream>>>(Qb, Kb, Vb, Ob);

  // 4) output projection (f32 straight to d_out)
  gemm_bf16_wmma<false><<<gemm_blocks, 256, 0, stream>>>(Ob, WoT, bo, nullptr, out);

  (void)in_sizes; (void)n_in; (void)out_size; (void)ws_size;
}
